// decoder_block_light_57148834840962
// MI455X (gfx1250) — matrix operand order repaired and checked by reading, not compiled, not run on hardware
//
#include <hip/hip_runtime.h>
#include <hip/hip_bf16.h>

typedef unsigned int u32;
typedef __attribute__((ext_vector_type(16))) __bf16 v16bf;
typedef __attribute__((ext_vector_type(8)))  float  v8f;
typedef __attribute__((ext_vector_type(4)))  u32    v4u;
typedef __attribute__((ext_vector_type(4)))  int    v4i;

union BFrag { v4u q[2]; v16bf v; };

#define ACT_NONE 0
#define ACT_RELU 1
#define ACT_ELU  2

#if __has_builtin(__builtin_amdgcn_global_load_async_to_lds_b128)
#define HAVE_ASYNC_LDS 1
#else
#define HAVE_ASYNC_LDS 0
#endif

__device__ __forceinline__ void wait_async_copies() {
#if HAVE_ASYNC_LDS
#if __has_builtin(__builtin_amdgcn_s_wait_asynccnt)
  __builtin_amdgcn_s_wait_asynccnt(0);
#else
  asm volatile("s_wait_asynccnt 0x0" ::: "memory");
#endif
#endif
}

__device__ __forceinline__ v8f wmma_bf16(BFrag a, BFrag b, v8f c) {
  // D(f32 16x16) = A(bf16 16x32) * B(bf16 32x16) + C
  return __builtin_amdgcn_wmma_f32_16x16x32_bf16(
      /*neg_a=*/false, a.v, /*neg_b=*/false, b.v,
      /*c_mod=*/(short)0, c, /*reuse_a=*/false, /*reuse_b=*/false);
}

#define LDS_ROW 36            // 32 dwords of weights + 4 pad dwords (bank spread)
#define LDS_TAP (64 * LDS_ROW)

// ---------------------------------------------------------------------------
// Sparse conv: out[j, :] = sum_k feats[nbr[k][j], :] @ W[k]   (Cin=Cout=64)
//   src : bf16 feature rows [N_src][64], viewed as u32 (2 bf16 per dword)
//   nbr : [27][n_dst] gather indices (-1 => zero row)
//   wt  : bf16 weights, per tap laid out [cout=64][cin=64] (transposed)
//   base: optional f32 residual add ([n_dst][64]), may alias out_f
//   out_f / out_b: optional f32 / bf16 outputs
// One wave computes a 32-row tile (2 x 16 M-tiles, 8 WMMA accumulators);
// 8 waves/block share per-tap weights staged in double-buffered LDS via
// async global->LDS copies (stage tap k+1 while computing tap k).
// ---------------------------------------------------------------------------
__global__ void __launch_bounds__(256)
k_spconv(const u32* __restrict__ src, const int* __restrict__ nbr,
         const u32* __restrict__ wt, const float* base,
         float* out_f, __hip_bfloat16* out_b, int n_dst, int act)
{
  __shared__ u32 lds_w[2][LDS_TAP];

  const int lane = threadIdx.x & 31;
  const int half = lane >> 4;      // K half (contiguous k = 16*half + s)
  const int nidx = lane & 15;      // row (A) / column (B,C,D) within a tile
  const int wave = threadIdx.x >> 5;
  const int row_base = (blockIdx.x * 8 + wave) * 32;

  // stage one tap's 8KB weight block into LDS buffer `bi` (bank-padded rows)
  auto stage = [&](int tap, int bi) {
    u32* g = const_cast<u32*>(wt) + (size_t)tap * 2048;
#pragma unroll
    for (int it = 0; it < 2; ++it) {
      int t = (int)threadIdx.x + it * 256;      // 0..511 16B chunks
      int r = t >> 3, q = t & 7;
#if HAVE_ASYNC_LDS
      __builtin_amdgcn_global_load_async_to_lds_b128(
          (__attribute__((address_space(1))) v4i*)(g + t * 4),
          (__attribute__((address_space(3))) v4i*)&lds_w[bi][r * LDS_ROW + q * 4],
          0, 0);
#else
      *(v4u*)&lds_w[bi][r * LDS_ROW + q * 4] = *(const v4u*)(g + t * 4);
#endif
    }
  };

  const v8f vzero = {0.f, 0.f, 0.f, 0.f, 0.f, 0.f, 0.f, 0.f};
  v8f acc[2][4];
#pragma unroll
  for (int mt = 0; mt < 2; ++mt)
#pragma unroll
    for (int c = 0; c < 4; ++c) acc[mt][c] = vzero;

  const v4u qzero = {0u, 0u, 0u, 0u};
  const int r0 = row_base + nidx;
  const int r1 = r0 + 16;

  stage(0, 0);                                  // prologue fill

  for (int k = 0; k < 27; ++k) {
    wait_async_copies();                        // my copies into buf[k&1] done
    __syncthreads();                            // all waves' copies published
    if (k + 1 < 27) stage(k + 1, (k + 1) & 1);  // overlap next tap's fill

    const u32* buf = lds_w[k & 1];
    const size_t nb = (size_t)k * (size_t)n_dst;
    int i0 = -1, i1 = -1;
    if (r0 < n_dst) i0 = nbr[nb + r0];
    if (r1 < n_dst) i1 = nbr[nb + r1];
    const u32* s0 = (i0 >= 0) ? (src + (size_t)i0 * 32) : (const u32*)0;
    const u32* s1 = (i1 >= 0) ? (src + (size_t)i1 * 32) : (const u32*)0;

#pragma unroll
    for (int f = 0; f < 2; ++f) {               // K slices 0..31, 32..63
      BFrag a0, a1;
      if (s0) {                                 // A 16x32, contiguous k = 16*half + s
        const v4u* p = (const v4u*)(s0 + f * 16 + half * 8);
        a0.q[0] = p[0]; a0.q[1] = p[1];
      } else { a0.q[0] = qzero; a0.q[1] = qzero; }
      if (s1) {
        const v4u* p = (const v4u*)(s1 + f * 16 + half * 8);
        a1.q[0] = p[0]; a1.q[1] = p[1];
      } else { a1.q[0] = qzero; a1.q[1] = qzero; }

#pragma unroll
      for (int c = 0; c < 4; ++c) {             // 4 output-column tiles of 16
        BFrag b;
        int off = (16 * c + nidx) * LDS_ROW + f * 16 + half * 8;
        b.q[0] = *(const v4u*)&buf[off];
        b.q[1] = *(const v4u*)&buf[off + 4];
        acc[0][c] = wmma_bf16(a0, b, acc[0][c]);  // B reused by both M tiles
        acc[1][c] = wmma_bf16(a1, b, acc[1][c]);
      }
    }
  }

  // epilogue: C/D layout -> lane holds (M = v + 8*half, N = nidx)
#pragma unroll
  for (int mt = 0; mt < 2; ++mt) {
#pragma unroll
    for (int c = 0; c < 4; ++c) {
#pragma unroll
      for (int v = 0; v < 8; ++v) {
        int row = row_base + mt * 16 + v + half * 8;
        if (row < n_dst) {
          int col = c * 16 + nidx;
          size_t e = (size_t)row * 64 + col;
          float val = acc[mt][c][v];
          if (base) val += base[e];
          if (act == ACT_RELU)     val = val > 0.f ? val : 0.f;
          else if (act == ACT_ELU) val = val > 0.f ? val : (__expf(val) - 1.f);
          if (out_f) out_f[e] = val;
          if (out_b) out_b[e] = __float2bfloat16(val);
        }
      }
    }
  }
}

// ---------------------------------------------------------------------------
// probs = p64 @ prob_w2   (N x 64) @ (64 x 16), w2t transposed [16][64] bf16
// ---------------------------------------------------------------------------
__global__ void __launch_bounds__(256)
k_prob2(const u32* __restrict__ p64, const u32* __restrict__ w2t,
        float* __restrict__ out, int n)
{
  const int lane = threadIdx.x & 31;
  const int half = lane >> 4;
  const int nidx = lane & 15;
  const int tile = blockIdx.x * 8 + (threadIdx.x >> 5);
  const int row_base = tile * 16;
  const int my_row = row_base + nidx;

  const v4u qzero = {0u, 0u, 0u, 0u};
  v8f acc = {0.f, 0.f, 0.f, 0.f, 0.f, 0.f, 0.f, 0.f};

#pragma unroll
  for (int f = 0; f < 2; ++f) {
    BFrag a;
    if (my_row < n) {
      const v4u* p = (const v4u*)(p64 + (size_t)my_row * 32 + f * 16 + half * 8);
      a.q[0] = p[0];
      a.q[1] = p[1];
    } else {
      a.q[0] = qzero;
      a.q[1] = qzero;
    }
    BFrag b;
    const u32* wr = w2t + nidx * 32 + f * 16 + half * 8;
    b.q[0] = *(const v4u*)wr;
    b.q[1] = *(const v4u*)(wr + 4);
    acc = wmma_bf16(a, b, acc);
  }

#pragma unroll
  for (int v = 0; v < 8; ++v) {
    int row = row_base + v + half * 8;
    if (row < n) out[(size_t)row * 16 + nidx] = acc[v];
  }
}

// x1 = x_feat @ w1 + f_feat ; h = x1 (f32 + bf16 copies). 0.65 GFLOP, scalar.
__global__ void k_x1(const float* __restrict__ xf, const float* __restrict__ ff,
                     const float* __restrict__ w1, float* __restrict__ x1,
                     float* __restrict__ hf, __hip_bfloat16* __restrict__ hb,
                     long long total)
{
  long long i = (long long)blockIdx.x * blockDim.x + threadIdx.x;
  if (i >= total) return;
  int row = (int)(i >> 6), c = (int)(i & 63);
  const float* xr = xf + (size_t)row * 128;
  float s = ff[i];
#pragma unroll 8
  for (int k = 0; k < 128; ++k) s += xr[k] * w1[k * 64 + c];
  x1[i] = s;
  hf[i] = s;
  hb[i] = __float2bfloat16(s);
}

// x3in_bf16 = bf16(x1 + h)
__global__ void k_addcvt(const float* __restrict__ a, const float* __restrict__ b,
                         __hip_bfloat16* __restrict__ o, long long total)
{
  long long i = (long long)blockIdx.x * blockDim.x + threadIdx.x;
  if (i >= total) return;
  o[i] = __float2bfloat16(a[i] + b[i]);
}

// transpose+convert weights: in [ntap][cin][cout] f32 -> out [ntap][cout][cin] bf16
__global__ void k_wt(const float* __restrict__ w, __hip_bfloat16* __restrict__ wt,
                     int ntap, int cin, int cout)
{
  long long i = (long long)blockIdx.x * blockDim.x + threadIdx.x;
  long long tot = (long long)ntap * cin * cout;
  if (i >= tot) return;
  int per = cin * cout;
  int k = (int)(i / per), r = (int)(i % per);
  int ci = r / cout, co = r % cout;
  wt[((size_t)k * cout + co) * cin + ci] = __float2bfloat16(w[i]);
}

static inline size_t rup(size_t x) { return (x + 255) & ~(size_t)255; }
static inline int ceil_div(long long a, long long b) { return (int)((a + b - 1) / b); }

extern "C" void kernel_launch(void* const* d_in, const int* in_sizes, int n_in_args,
                              void* d_out, int out_size, void* d_ws, size_t ws_size,
                              hipStream_t stream) {
  (void)n_in_args; (void)out_size; (void)ws_size;
  const float* x_feat  = (const float*)d_in[0];
  const float* f_feat  = (const float*)d_in[1];
  const float* w1      = (const float*)d_in[2];
  const float* res_w   = (const float*)d_in[3];
  const float* up_w    = (const float*)d_in[4];
  const float* upper_w = (const float*)d_in[5];
  const float* prob_w1 = (const float*)d_in[6];
  const float* prob_w2 = (const float*)d_in[7];
  const int*   nbr_in  = (const int*)d_in[8];
  const int*   up_nbr  = (const int*)d_in[9];
  const int*   nbr_out = (const int*)d_in[10];

  const int n_in  = in_sizes[0] / 128;   // 40000
  const int n_out = in_sizes[9] / 27;    // runtime output point count
  const long long nin64  = (long long)n_in * 64;
  const long long nout64 = (long long)n_out * 64;

  // ---- carve workspace ----
  char* p = (char*)d_ws;
  float* x1f = (float*)p;                     p += rup(nin64 * 4);
  float* hf  = (float*)p;                     p += rup(nin64 * 4);
  __hip_bfloat16* hb    = (__hip_bfloat16*)p; p += rup(nin64 * 2);
  __hip_bfloat16* tb    = (__hip_bfloat16*)p; p += rup(nin64 * 2);
  __hip_bfloat16* x3b   = (__hip_bfloat16*)p; p += rup(nin64 * 2);
  __hip_bfloat16* reswt = (__hip_bfloat16*)p; p += rup((size_t)162 * 4096 * 2);
  __hip_bfloat16* upwt  = (__hip_bfloat16*)p; p += rup((size_t)27 * 4096 * 2);
  __hip_bfloat16* uppwt = (__hip_bfloat16*)p; p += rup((size_t)27 * 4096 * 2);
  __hip_bfloat16* pr1wt = (__hip_bfloat16*)p; p += rup((size_t)27 * 4096 * 2);
  __hip_bfloat16* w2t   = (__hip_bfloat16*)p; p += rup((size_t)16 * 64 * 2);
  __hip_bfloat16* yb    = (__hip_bfloat16*)p; p += rup(nout64 * 2);
  __hip_bfloat16* x3ob  = (__hip_bfloat16*)p; p += rup(nout64 * 2);
  __hip_bfloat16* p64b  = (__hip_bfloat16*)p; p += rup(nout64 * 2);

  float* out_x3    = (float*)d_out;                  // [n_out][64]
  float* out_probs = (float*)d_out + nout64;         // [n_out][16]

  const dim3 blk(256);

  // ---- weight transpose+convert to bf16 [cout][cin] ----
  k_wt<<<ceil_div((long long)162 * 4096, 256), blk, 0, stream>>>(res_w,   reswt, 162, 64, 64);
  k_wt<<<ceil_div((long long)27  * 4096, 256), blk, 0, stream>>>(up_w,    upwt,  27,  64, 64);
  k_wt<<<ceil_div((long long)27  * 4096, 256), blk, 0, stream>>>(upper_w, uppwt, 27,  64, 64);
  k_wt<<<ceil_div((long long)27  * 4096, 256), blk, 0, stream>>>(prob_w1, pr1wt, 27,  64, 64);
  k_wt<<<ceil_div((long long)1   * 1024, 256), blk, 0, stream>>>(prob_w2, w2t,   1,   64, 16);

  // ---- x1 = x_feat @ w1 + f_feat ; h = x1 ----
  k_x1<<<ceil_div(nin64, 256), blk, 0, stream>>>(x_feat, f_feat, w1, x1f, hf, hb, nin64);

  // 32 rows per wave, 8 waves per block
  const int grid_in  = ceil_div(((long long)n_in  + 31) / 32, 8);
  const int grid_out = ceil_div(((long long)n_out + 31) / 32, 8);

  // ---- residual blocks: h += convB(relu(convA(h))) ----
  for (int r = 0; r < 3; ++r) {
    const u32* wa = (const u32*)reswt + (size_t)(r * 2 + 0) * 27 * 2048;
    const u32* wb = (const u32*)reswt + (size_t)(r * 2 + 1) * 27 * 2048;
    k_spconv<<<grid_in, blk, 0, stream>>>((const u32*)hb, nbr_in, wa,
                                          nullptr, nullptr, tb, n_in, ACT_RELU);
    k_spconv<<<grid_in, blk, 0, stream>>>((const u32*)tb, nbr_in, wb,
                                          hf, hf, hb, n_in, ACT_NONE);
  }

  // ---- x3_in = x1 + h (bf16 for upconv gathers) ----
  k_addcvt<<<ceil_div(nin64, 256), blk, 0, stream>>>(x1f, hf, x3b, nin64);

  // ---- transpose conv stride2->1: N_in -> N_out ----
  k_spconv<<<grid_out, blk, 0, stream>>>((const u32*)x3b, up_nbr, (const u32*)upwt,
                                         nullptr, nullptr, yb, n_out, ACT_NONE);

  // ---- to_upper conv + ELU -> first output ----
  k_spconv<<<grid_out, blk, 0, stream>>>((const u32*)yb, nbr_out, (const u32*)uppwt,
                                         nullptr, out_x3, x3ob, n_out, ACT_ELU);

  // ---- prob conv (3x3x3) then 1x1 projection -> second output ----
  k_spconv<<<grid_out, blk, 0, stream>>>((const u32*)x3ob, nbr_out, (const u32*)pr1wt,
                                         nullptr, nullptr, p64b, n_out, ACT_NONE);
  k_prob2<<<ceil_div(((long long)n_out + 15) / 16, 8), blk, 0, stream>>>(
      (const u32*)p64b, (const u32*)w2t, out_probs, n_out);
}
